// LSTM_39848706573846
// MI455X (gfx1250) — matrix operand order repaired and checked by reading, not compiled, not run on hardware
//
#include <hip/hip_runtime.h>
#include <hip/hip_bf16.h>
#include <stdint.h>

// LSTM acceptor: B=128, T=2048, E=128, H=256 (4H = 1024)
// MI455X / gfx1250 (wave32, WMMA) strategy:
//  - recurrence is independent per batch row -> 8 persistent WGs x 16 samples,
//    zero inter-WG sync; each WG scans T=2048 locally.
//  - 16 waves/WG; wave w owns hidden units [16w,16w+16) across all 4 gates,
//    so c/h/last stay in WMMA C/D-layout VGPRs the whole scan.
//  - f16 WMMA 16x16x32, f32 accumulate. Weights repacked once into a merged
//    B-fragment stream (W_ih ktiles 0..3, W_hh ktiles 4..11) in d_ws (768 KB,
//    L2-resident); base pointer is laundered per step (kills LICM -> no
//    scratch spills) then cast back to addrspace(1) so the loads lower to
//    global_load_b128 (LOADcnt only, decoupled from the DS pipe).
//  - h written per step directly in A-fragment f16 layout in LDS (double
//    buffered); x_t staged one step ahead -> 2 barriers per step.

#define Bsz  128
#define Tlen 2048
#define Edim 128
#define Hdim 256

typedef __attribute__((ext_vector_type(16))) _Float16     v16h;
typedef __attribute__((ext_vector_type(8)))  float        v8f;
typedef __attribute__((ext_vector_type(2)))  __fp16       fp16x2;
typedef __attribute__((ext_vector_type(4)))  unsigned int u32x4;   // plain vector (no class op=)

typedef const __attribute__((address_space(1))) u32x4* gvp;        // global-AS ptr

union Frag { v16h v; u32x4 q[2]; unsigned int d[8]; };

__device__ __forceinline__ unsigned int pk2(float lo, float hi) {
  union { fp16x2 p; unsigned int u; } cv;
  cv.p = __builtin_amdgcn_cvt_pkrtz(lo, hi);   // v_cvt_pk_rtz_f16_f32
  return cv.u;
}
__device__ __forceinline__ float fexp2f(float x) { return __builtin_amdgcn_exp2f(x); }
__device__ __forceinline__ float frcpf (float x) { return __builtin_amdgcn_rcpf(x); }
__device__ __forceinline__ float sigf  (float x) { return frcpf(1.0f + fexp2f(-1.44269504f * x)); }
__device__ __forceinline__ float tanhfast(float x) { return 2.0f * frcpf(1.0f + fexp2f(-2.88539008f * x)) - 1.0f; }

// merged weight stream: 64 ntiles x 12 ktiles (4 from W_ih, 8 from W_hh)
// dword index = ((nt*12 + kt)*32 + lane)*8 + v   -> 32B contiguous per lane
#define KT_ALL 12
#define WP_DW  (64 * KT_ALL * 32 * 8)   // 196608 dwords = 768 KB

// ---------------------------------------------------------------------------
// Prologue: f32 weights -> f16 WMMA B-fragments. B frag (K=32 x N=16, 16-bit):
// N = lane%16; element s = 2v (lo), 2v+1 (hi) holds K = 8h+s (s<8), 16+8h+(s-8) (s>=8), h = lane/16.
// ---------------------------------------------------------------------------
__global__ __launch_bounds__(256) void lstm_repack(const float* __restrict__ W_ih,
                                                   const float* __restrict__ W_hh,
                                                   unsigned int* __restrict__ wp) {
  int id = blockIdx.x * 256 + threadIdx.x;     // < WP_DW
  int v    = id & 7;
  int lane = (id >> 3) & 31;
  int tmp  = id >> 8;
  int kt   = tmp % KT_ALL;
  int nt   = tmp / KT_ALL;
  int n0   = nt * 16 + (lane & 15);
  int hh   = lane >> 4;
  int s0   = 2 * v;
  int kl   = (s0 < 8) ? (8 * hh + s0) : (16 + 8 * hh + (s0 - 8));
  float lo, hi;
  if (kt < 4) {            // W_ih ktile
    int k = kt * 32 + kl;
    lo = W_ih[n0 * Edim + k];
    hi = W_ih[n0 * Edim + k + 1];
  } else {                 // W_hh ktile
    int k = (kt - 4) * 32 + kl;
    lo = W_hh[n0 * Hdim + k];
    hi = W_hh[n0 * Hdim + k + 1];
  }
  wp[id] = pk2(lo, hi);
}

// ---------------------------------------------------------------------------
// Persistent scan kernel: 8 blocks x 512 threads (16 waves).
// ---------------------------------------------------------------------------
__global__ __launch_bounds__(512) void lstm_scan(const float* __restrict__ seq,
                                                 const int*   __restrict__ lens,
                                                 const float* __restrict__ b_ih,
                                                 const float* __restrict__ b_hh,
                                                 const u32x4* __restrict__ wp,
                                                 float* __restrict__ out) {
  __shared__ float xs[2][16][Edim + 4];   // x_t staging, f32, padded rows
  __shared__ u32x4 hA4[2][512];           // h in A-frag f16 layout: dword = kt*256 + lane*8 + v

  const int tid   = threadIdx.x;
  const int lane  = tid & 31;
  const int w     = tid >> 5;            // wave 0..15
  const int base  = blockIdx.x * 16;     // batch tile base
  const int N0    = lane & 15;           // N within tile; also A-matrix row M
  const int halfD = lane >> 4;           // C/D (and A) lane half

  // --- per-lane constants -------------------------------------------------
  const int hid  = w * 16 + N0;          // hidden unit this lane owns in C/D
  // inverse A-frag mapping for storing h[M][hid] as f16:
  const int kt_h   = hid >> 5;
  const int khid   = hid & 31;
  const int grp    = khid >> 3;
  const int half_h = grp & 1;
  const int v_h    = (grp >> 1) * 4 + ((khid & 7) >> 1);
  const int b_h    = khid & 1;

  float biasv[4];
#pragma unroll
  for (int g = 0; g < 4; ++g) {
    int n = g * Hdim + hid;
    biasv[g] = b_ih[n] + b_hh[n];
  }
  int lenv[8];
#pragma unroll
  for (int r = 0; r < 8; ++r) lenv[r] = lens[base + r + 8 * halfD];

  // --- init ---------------------------------------------------------------
  { u32x4 z = {0u, 0u, 0u, 0u}; hA4[0][tid] = z; }   // h_0 = 0
  v8f c = {};  v8f last = {};

  // x staging: thread -> (row = w, 4 consecutive floats), fully coalesced
  const int xrow = tid >> 5;
  const int xcol = (tid & 31) * 4;
  const float* xbase = seq + (size_t)(base + xrow) * Tlen * Edim + xcol;

  float4 x0 = *(const float4*)xbase;     // t = 0
  xs[0][xrow][xcol + 0] = x0.x;
  xs[0][xrow][xcol + 1] = x0.y;
  xs[0][xrow][xcol + 2] = x0.z;
  xs[0][xrow][xcol + 3] = x0.w;

#pragma unroll 1
  for (int t = 0; t < Tlen; ++t) {
    const int p = t & 1;

    // Launder the weight pointer each step (kills LICM/CSE of the B-frag
    // loads), then cast back to global AS so they lower to global_load_b128.
    const u32x4* wl = wp;
    asm volatile("" : "+s"(wl));
    gvp wbase = (gvp)(const void*)wl;

    __syncthreads();   // xs[p], hA4[p] (written during t-1) now visible

    // prefetch x_{t+1} early: a full step of WMMA work hides HBM latency
    const bool havenext = (t + 1) < Tlen;
    float4 xn = {0.f, 0.f, 0.f, 0.f};
    if (havenext) xn = *(const float4*)(xbase + (size_t)(t + 1) * Edim);

    // ---- build A fragments: af[0..3] = x_t, af[4..11] = h_t -----------
    Frag af[KT_ALL];
#pragma unroll
    for (int kt = 0; kt < 4; ++kt) {
#pragma unroll
      for (int v = 0; v < 8; ++v) {
        int kk = kt * 32 + ((v >> 2) * 2 + halfD) * 8 + (v & 3) * 2;
        af[kt].d[v] = pk2(xs[p][N0][kk], xs[p][N0][kk + 1]);
      }
    }
    const u32x4* hsrc = &hA4[p][0];
#pragma unroll
    for (int kt = 0; kt < 8; ++kt) {     // already in A-frag layout: 2x b128
      int i4 = kt * 64 + lane * 2;
      af[4 + kt].q[0] = hsrc[i4];
      af[4 + kt].q[1] = hsrc[i4 + 1];
    }

    __syncthreads();   // all reads of xs[p]/hA4[p] done; safe to refill below

    // ---- gates = x_t*W_ih^T + h*W_hh^T (f32 accum), B double-buffered --
    v8f acc[4];
#pragma unroll
    for (int g = 0; g < 4; ++g) {
      gvp src = wbase + (size_t)(g * 16 + w) * (KT_ALL * 64) + lane * 2;
      Frag bfa, bfb;
      bfa.q[0] = src[0];
      bfa.q[1] = src[1];
      v8f a = {};
#pragma unroll
      for (int kt = 0; kt < KT_ALL; ++kt) {
        if (kt + 1 < KT_ALL) {           // prefetch next B frag
          bfb.q[0] = src[(kt + 1) * 64 + 0];
          bfb.q[1] = src[(kt + 1) * 64 + 1];
        }
        a = __builtin_amdgcn_wmma_f32_16x16x32_f16(false, af[kt].v, false, bfa.v,
                                                   (short)0, a, false, false);
        bfa = bfb;
      }
      acc[g] = a;
    }

    // ---- elementwise: c,h,last in C/D register layout -----------------
    _Float16* hdst = (_Float16*)&hA4[1 - p][0];
#pragma unroll
    for (int r = 0; r < 8; ++r) {
      float gi = sigf    (acc[0][r] + biasv[0]);
      float gf = sigf    (acc[1][r] + biasv[1]);
      float gg = tanhfast(acc[2][r] + biasv[2]);
      float go = sigf    (acc[3][r] + biasv[3]);
      float cn = gf * c[r] + gi * gg;
      c[r] = cn;
      float hv = go * tanhfast(cn);
      last[r] = (lenv[r] > t) ? hv : last[r];
      int M  = r + 8 * halfD;
      int us = (kt_h * 256 + (16 * half_h + M) * 8 + v_h) * 2 + b_h;
      hdst[us] = (_Float16)hv;          // h_{t+1} in A-frag layout
    }

    // stage x_{t+1} into the other buffer
    if (havenext) {
      xs[1 - p][xrow][xcol + 0] = xn.x;
      xs[1 - p][xrow][xcol + 1] = xn.y;
      xs[1 - p][xrow][xcol + 2] = xn.z;
      xs[1 - p][xrow][xcol + 3] = xn.w;
    }
  }

  // ---- write last hidden state: out[B,H] f32 ----------------------------
#pragma unroll
  for (int r = 0; r < 8; ++r)
    out[(size_t)(base + r + 8 * halfD) * Hdim + hid] = last[r];
}

// ---------------------------------------------------------------------------
extern "C" void kernel_launch(void* const* d_in, const int* in_sizes, int n_in,
                              void* d_out, int out_size, void* d_ws, size_t ws_size,
                              hipStream_t stream) {
  (void)in_sizes; (void)n_in; (void)out_size; (void)ws_size;
  const float* seq  = (const float*)d_in[0];
  const int*   lens = (const int*)  d_in[1];
  const float* W_ih = (const float*)d_in[2];
  const float* W_hh = (const float*)d_in[3];
  const float* b_ih = (const float*)d_in[4];
  const float* b_hh = (const float*)d_in[5];

  unsigned int* wp = (unsigned int*)d_ws;   // 768 KB packed weight fragments

  lstm_repack<<<WP_DW / 256, 256, 0, stream>>>(W_ih, W_hh, wp);
  lstm_scan<<<Bsz / 16, 512, 0, stream>>>(seq, lens, b_ih, b_hh,
                                          (const u32x4*)wp, (float*)d_out);
}
